// GCNLPA_50646254354789
// MI455X (gfx1250) — matrix operand order repaired and checked by reading, not compiled, not run on hardware
//
#include <hip/hip_runtime.h>
#include <hip/hip_bf16.h>
#include <stdint.h>

#define NN   50000     // nodes
#define NE   1600000   // edges
#define CC   64        // classes
#define DIN  256
#define DH   128
#define KLP  4

typedef __attribute__((ext_vector_type(16))) __bf16 v16bf;
typedef __attribute__((ext_vector_type(8)))  float  v8f;

static __device__ __forceinline__ unsigned short f2bf(float f) {
  unsigned u = __builtin_bit_cast(unsigned, f);
  u += 0x7FFFu + ((u >> 16) & 1u);        // round-to-nearest-even
  return (unsigned short)(u >> 16);
}

// ---------------- prep kernels ----------------

__global__ void k_init_deg(float* deg) {
  int i = blockIdx.x * blockDim.x + threadIdx.x;
  if (i < NN) deg[i] = 1.0f;              // self-loop weight 1
}

__global__ void k_ew_deg(const float* __restrict__ edge_w, const int* __restrict__ dst,
                         float* __restrict__ ew, float* __restrict__ deg) {
  int e = blockIdx.x * blockDim.x + threadIdx.x;
  if (e >= NE) return;
  float w = 1.0f / (1.0f + __expf(-edge_w[e]));   // sigmoid
  ew[e] = w;
  atomicAdd(&deg[dst[e]], w);
}

__global__ void k_rsqrt(float* d) {
  int i = blockIdx.x * blockDim.x + threadIdx.x;
  if (i < NN) d[i] = rsqrtf(d[i]);
}

__global__ void k_f32_to_bf16(const float* __restrict__ in, unsigned short* __restrict__ out, int n) {
  int i = blockIdx.x * blockDim.x + threadIdx.x;
  if (i < n) out[i] = f2bf(in[i]);
}

// W [K,N] row-major f32 -> Wt [N,K] bf16 (so B fragments are contiguous per lane)
__global__ void k_transpose_bf16(const float* __restrict__ W, unsigned short* __restrict__ Wt,
                                 int K, int Nc) {
  int i = blockIdx.x * blockDim.x + threadIdx.x;
  if (i >= K * Nc) return;
  int k = i / Nc, n = i % Nc;
  Wt[n * K + k] = f2bf(W[i]);
}

// ---------------- WMMA GEMM: C[M,NOUT] = A[M,K] (bf16) * Bt[NOUT,K]^T (bf16) ----------------
// One wave per 16x16 output tile. Per k-step: 4x global_load_b128 + 1x v_wmma_f32_16x16x32_bf16.
template<int KDIM, int NOUT>
__global__ void k_gemm_wmma(const unsigned short* __restrict__ Abf,
                            const unsigned short* __restrict__ Btbf,
                            float* __restrict__ Cout, int M) {
  int wave = (blockIdx.x * blockDim.x + threadIdx.x) >> 5;
  int lane = threadIdx.x & 31;
  const int ntiles = NOUT / 16;
  int mtile = wave / ntiles;
  int ntile = wave % ntiles;
  if (mtile * 16 >= M) return;

  int row = mtile * 16 + (lane & 15);
  int col = ntile * 16 + (lane & 15);
  // A layout (16-bit 16x32): lanes 0-15 hold K=[0..7]+[16..23], lanes 16-31 K=[8..15]+[24..31]
  int akb = (lane < 16) ? 0 : 8;
  // B layout (16-bit 32x16): lanes 0-15 hold K=[0..7]+[16..23], lanes 16-31 K=[8..15]+[24..31]
  int bkb = (lane < 16) ? 0 : 8;
  const unsigned short* arow = Abf + (size_t)row * KDIM + akb;
  const unsigned short* brow = Btbf + (size_t)col * KDIM + bkb;

  v8f acc = {};
#pragma unroll
  for (int k0 = 0; k0 < KDIM; k0 += 32) {
    union { v16bf v; uint4 q[2]; } a, b;
    a.q[0] = *reinterpret_cast<const uint4*>(arow + k0);        // K = akb .. akb+7
    a.q[1] = *reinterpret_cast<const uint4*>(arow + k0 + 16);   // K = akb+16 .. akb+23
    b.q[0] = *reinterpret_cast<const uint4*>(brow + k0);        // K = bkb .. bkb+7
    b.q[1] = *reinterpret_cast<const uint4*>(brow + k0 + 16);   // K = bkb+16 .. bkb+23
    acc = __builtin_amdgcn_wmma_f32_16x16x32_bf16(
        false, a.v, false, b.v, (short)0, acc, false, false);
  }
  // C layout: VGPR r -> M = r (lanes 0-15) / r+8 (lanes 16-31), N = lane&15
  int rbase = mtile * 16 + ((lane < 16) ? 0 : 8);
#pragma unroll
  for (int r = 0; r < 8; ++r)
    Cout[(size_t)(rbase + r) * NOUT + col] = acc[r];
}

// ---------------- aggregation ----------------

// agg[i,:] = dinv[i]^2 * h[i,:] + bias  (self-loop term + bias), float4 vectorized
template<int D>
__global__ void k_agg_init(const float* __restrict__ h, const float* __restrict__ dinv,
                           const float* __restrict__ bias, float* __restrict__ agg) {
  const int Q = D / 4;
  int i = blockIdx.x * blockDim.x + threadIdx.x;
  if (i >= NN * Q) return;
  int node = i / Q, c = i % Q;
  float s = dinv[node]; s *= s;
  float4 hv = reinterpret_cast<const float4*>(h)[i];
  float4 bv = reinterpret_cast<const float4*>(bias)[c];
  float4 r;
  r.x = s * hv.x + bv.x; r.y = s * hv.y + bv.y;
  r.z = s * hv.z + bv.z; r.w = s * hv.w + bv.w;
  reinterpret_cast<float4*>(agg)[i] = r;
}

// agg[sidx[e],:] += w_e * h[gidx[e],:]   (dinv!=null: w = dinv[g]*ew*dinv[s]; else w = ew)
// D/4 lanes per edge, float4 gather + 4 f32 atomics (L2-resident targets).
template<int D>
__global__ void k_edge_scatter(const int* __restrict__ gidx, const int* __restrict__ sidx,
                               const float* __restrict__ ew, const float* __restrict__ dinv,
                               const float* __restrict__ hin, float* __restrict__ agg) {
  const int LPE = D / 4;
  long long t = (long long)blockIdx.x * blockDim.x + threadIdx.x;
  int e = (int)(t / LPE);
  int c = (int)(t % LPE);
  if (e >= NE) return;
  int g = gidx[e], s = sidx[e];
  float w = ew[e];
  if (dinv) w *= dinv[g] * dinv[s];
  float4 hv = reinterpret_cast<const float4*>(hin + (size_t)g * D)[c];
  float* ap = agg + (size_t)s * D + c * 4;
  atomicAdd(ap + 0, w * hv.x);
  atomicAdd(ap + 1, w * hv.y);
  atomicAdd(ap + 2, w * hv.z);
  atomicAdd(ap + 3, w * hv.w);
}

__global__ void k_relu_to_bf16(const float* __restrict__ in, unsigned short* __restrict__ out, int n) {
  int i = blockIdx.x * blockDim.x + threadIdx.x;
  if (i < n) out[i] = f2bf(fmaxf(in[i], 0.0f));
}

// ---------------- row ops (one wave32 per row of 64) ----------------

__global__ void k_softmax64(const float* __restrict__ in, float* __restrict__ out) {
  int wave = (blockIdx.x * blockDim.x + threadIdx.x) >> 5;
  int lane = threadIdx.x & 31;
  if (wave >= NN) return;
  const float* r = in + (size_t)wave * CC;
  float a = r[lane], b = r[lane + 32];
  float m = fmaxf(a, b);
#pragma unroll
  for (int o = 16; o; o >>= 1) m = fmaxf(m, __shfl_xor(m, o, 32));
  float ea = __expf(a - m), eb = __expf(b - m);
  float s = ea + eb;
#pragma unroll
  for (int o = 16; o; o >>= 1) s += __shfl_xor(s, o, 32);
  float inv = 1.0f / s;
  out[(size_t)wave * CC + lane]      = ea * inv;
  out[(size_t)wave * CC + lane + 32] = eb * inv;
}

__global__ void k_onehot(const int* __restrict__ y, float* __restrict__ L) {
  int i = blockIdx.x * blockDim.x + threadIdx.x;
  if (i >= NN * CC) return;
  int node = i >> 6, c = i & 63;
  L[i] = (y[node] == c) ? 1.0f : 0.0f;
}

__global__ void k_copy4(const float4* __restrict__ a, float4* __restrict__ b, int n) {
  int i = blockIdx.x * blockDim.x + threadIdx.x;
  if (i < n) b[i] = a[i];
}

__global__ void k_rownorm64(const float* __restrict__ in, float* __restrict__ out) {
  int wave = (blockIdx.x * blockDim.x + threadIdx.x) >> 5;
  int lane = threadIdx.x & 31;
  if (wave >= NN) return;
  const float* r = in + (size_t)wave * CC;
  float a = r[lane], b = r[lane + 32];
  float s = a * a + b * b;
#pragma unroll
  for (int o = 16; o; o >>= 1) s += __shfl_xor(s, o, 32);
  float inv = 1.0f / fmaxf(sqrtf(s), 1e-12f);
  out[(size_t)wave * CC + lane]      = a * inv;
  out[(size_t)wave * CC + lane + 32] = b * inv;
}

// ---------------- launch ----------------

static inline int cdiv(long long a, long long b) { return (int)((a + b - 1) / b); }

extern "C" void kernel_launch(void* const* d_in, const int* in_sizes, int n_in,
                              void* d_out, int out_size, void* d_ws, size_t ws_size,
                              hipStream_t stream) {
  const float* x      = (const float*)d_in[0];
  const int*   ei     = (const int*)d_in[1];
  const int*   src    = ei;            // edge_index[0]
  const int*   dst    = ei + NE;       // edge_index[1]
  const int*   y      = (const int*)d_in[2];
  const float* edge_w = (const float*)d_in[3];
  const float* W1     = (const float*)d_in[4];
  const float* b1     = (const float*)d_in[5];
  const float* W2     = (const float*)d_in[6];
  const float* b2     = (const float*)d_in[7];

  float* out_sm = (float*)d_out;                       // softmax [N,64]
  float* out_lb = (float*)d_out + (size_t)NN * CC;     // labels  [N,64]

  // workspace carve-out (256B aligned regions)
  char* w = (char*)d_ws;
  auto carve = [&](size_t bytes) { char* p = w; w += (bytes + 255) & ~(size_t)255; return p; };
  float*          ew   = (float*)carve((size_t)NE * 4);
  float*          dinv = (float*)carve((size_t)NN * 4);
  unsigned short* xh   = (unsigned short*)carve((size_t)NN * DIN * 2);  // bf16 x; reused as h1h
  unsigned short* W1t  = (unsigned short*)carve((size_t)DH * DIN * 2);
  unsigned short* W2t  = (unsigned short*)carve((size_t)CC * DH * 2);
  float*          h1   = (float*)carve((size_t)NN * DH * 4);            // reused: h2 | labelsB
  float*          agg1 = (float*)carve((size_t)NN * DH * 4);            // reused: agg2 | labelsA
  unsigned short* h1h  = xh;                       // xh dead after GEMM1
  float* h2      = h1;                             // h1 dead after layer-1 scatter
  float* labelsB = h1 + (size_t)NN * CC;
  float* agg2    = agg1;                           // agg1 dead after relu conversion
  float* labelsA = agg1 + (size_t)NN * CC;

  const int B = 256;

  // --- degree / edge weights ---
  k_init_deg<<<cdiv(NN, B), B, 0, stream>>>(dinv);                     // dinv holds deg first
  k_ew_deg<<<cdiv(NE, B), B, 0, stream>>>(edge_w, dst, ew, dinv);
  k_rsqrt<<<cdiv(NN, B), B, 0, stream>>>(dinv);

  // --- precision prep ---
  k_f32_to_bf16<<<cdiv((long long)NN * DIN, B), B, 0, stream>>>(x, xh, NN * DIN);
  k_transpose_bf16<<<cdiv(DIN * DH, B), B, 0, stream>>>(W1, W1t, DIN, DH);
  k_transpose_bf16<<<cdiv(DH * CC, B), B, 0, stream>>>(W2, W2t, DH, CC);

  // --- layer 1: h1 = x @ W1 (WMMA), then agg1 = norm-propagated + bias, relu->bf16 ---
  {
    long long waves = (long long)(NN / 16) * (DH / 16);
    k_gemm_wmma<DIN, DH><<<cdiv(waves * 32, B), B, 0, stream>>>(xh, W1t, h1, NN);
  }
  k_agg_init<DH><<<cdiv((long long)NN * DH / 4, B), B, 0, stream>>>(h1, dinv, b1, agg1);
  k_edge_scatter<DH><<<cdiv((long long)NE * (DH / 4), B), B, 0, stream>>>(src, dst, ew, dinv, h1, agg1);
  k_relu_to_bf16<<<cdiv((long long)NN * DH, B), B, 0, stream>>>(agg1, h1h, NN * DH);

  // --- layer 2: h2 = relu1 @ W2 (WMMA), agg2 = propagated + bias, softmax ---
  {
    long long waves = (long long)(NN / 16) * (CC / 16);
    k_gemm_wmma<DH, CC><<<cdiv(waves * 32, B), B, 0, stream>>>(h1h, W2t, h2, NN);
  }
  k_agg_init<CC><<<cdiv((long long)NN * CC / 4, B), B, 0, stream>>>(h2, dinv, b2, agg2);
  k_edge_scatter<CC><<<cdiv((long long)NE * (CC / 4), B), B, 0, stream>>>(src, dst, ew, dinv, h2, agg2);
  k_softmax64<<<cdiv((long long)NN * 32, B), B, 0, stream>>>(agg2, out_sm);

  // --- label propagation: L <- scatter_src(ew * L[dst]) + L, 4 times ---
  k_onehot<<<cdiv((long long)NN * CC, B), B, 0, stream>>>(y, labelsA);
  float* cur = labelsA;
  float* nxt = labelsB;
  for (int k = 0; k < KLP; ++k) {
    k_copy4<<<cdiv((long long)NN * CC / 4, B), B, 0, stream>>>((const float4*)cur, (float4*)nxt, NN * CC / 4);
    k_edge_scatter<CC><<<cdiv((long long)NE * (CC / 4), B), B, 0, stream>>>(dst, src, ew, nullptr, cur, nxt);
    float* t = cur; cur = nxt; nxt = t;
  }
  k_rownorm64<<<cdiv((long long)NN * 32, B), B, 0, stream>>>(cur, out_lb);
}
